// Patient2Vec2_26353919328296
// MI455X (gfx1250) — compile-verified, matrix operand order checked by reading, not run on hardware
//
#include <hip/hip_runtime.h>
#include <hip/hip_bf16.h>

typedef __attribute__((ext_vector_type(16))) __bf16 bf16x16;
typedef __attribute__((ext_vector_type(8)))  float  f32x8;

// ---------- helpers ----------------------------------------------------------

__device__ __forceinline__ unsigned short f2bf(float x) {
  unsigned u = __float_as_uint(x);
  return (unsigned short)((u + 0x7FFFu + ((u >> 16) & 1u)) >> 16);  // RNE
}

union BF16Frag { bf16x16 v; unsigned u[8]; };

// Load a 16x32 bf16 WMMA fragment for one lane. rowptr points at the start of
// this lane's row (A: M-row, B: N-row of W so that B = W^T column-major).
// Per CDNA5 ISA 7.12.2 (16-bit A-matrix 16x32): VGPR q holds K pair
//   k = k0 + (q<4 ? 0 : 16) + (q&3)*2 + 8*(lane>>4)
// (adjacent pairs coalesce into two b128 loads per fragment)
__device__ __forceinline__ bf16x16 load_frag(const unsigned short* rowptr,
                                             int hi, int k0) {
  BF16Frag f;
#pragma unroll
  for (int q = 0; q < 8; ++q) {
    int k = k0 + ((q < 4) ? 0 : 16) + (q & 3) * 2 + (hi << 3);
    f.u[q] = *(const unsigned*)(rowptr + k);
  }
  return f.v;
}

// ---------- fp32 -> bf16 weight cast ----------------------------------------

__global__ void cast_bf16_kernel(const float* __restrict__ src,
                                 unsigned short* __restrict__ dst, int n) {
  int i = blockIdx.x * blockDim.x + threadIdx.x;
  if (i < n) dst[i] = f2bf(src[i]);
}

// ---------- stage 1: attention conv + embedding ------------------------------
// one block per token (b*64+s); inputs tile 32x256 staged in LDS

__global__ void __launch_bounds__(256) attn_conv_kernel(
    const float* __restrict__ inputs,    // [2048][32][256]
    const float* __restrict__ conv_w,    // [4][256]
    const float* __restrict__ conv_b,    // [4]
    float* __restrict__ conv_wts_out,    // [2048][4][32]
    unsigned short* __restrict__ emb) {  // [2048][1024] bf16
  __shared__ float tileIn[32][256];
  __shared__ float wts[4][32];
  int tok = blockIdx.x;
  int tid = threadIdx.x;
  const float* ip = inputs + (size_t)tok * 8192;
#pragma unroll
  for (int r = 0; r < 32; ++r) tileIn[r][tid] = ip[r * 256 + tid];
  __syncthreads();
  if (tid < 128) {
    int f = tid >> 5, p = tid & 31;
    float s = conv_b[f];
    const float* w = conv_w + f * 256;
    for (int i = 0; i < 256; ++i) s += tileIn[p][i] * w[i];
    wts[f][p] = s;
  }
  __syncthreads();
  if (tid < 4) {  // softmax over p
    float m = -3.4e38f;
    for (int p = 0; p < 32; ++p) m = fmaxf(m, wts[tid][p]);
    float sum = 0.f;
    for (int p = 0; p < 32; ++p) { float e = __expf(wts[tid][p] - m); wts[tid][p] = e; sum += e; }
    float inv = 1.f / sum;
    for (int p = 0; p < 32; ++p) wts[tid][p] *= inv;
  }
  __syncthreads();
  if (tid < 128) {
    int f = tid >> 5, p = tid & 31;
    conv_wts_out[(size_t)tok * 128 + f * 32 + p] = wts[f][p];
  }
  int f = tid >> 6, i0 = (tid & 63) * 4;
#pragma unroll
  for (int ii = 0; ii < 4; ++ii) {
    int i = i0 + ii;
    float s = 0.f;
    for (int p = 0; p < 32; ++p) s += wts[f][p] * tileIn[p][i];
    emb[(size_t)tok * 1024 + f * 256 + i] = f2bf(s);
  }
}

// ---------- stage 2a: gx = x @ wih^T + bih (both directions) -----------------
// M=2048 tokens, N=1536 gates, K=1024.
// One wave computes a 16x64 strip (4 N-tiles): A fragment loaded once per
// K-step, reused across 4 WMMAs, with the next K-step's fragments prefetched
// (double-buffered) so WMMAs overlap outstanding loads.

__global__ void __launch_bounds__(256) gemm_gx_kernel(
    const unsigned short* __restrict__ x,    // [2048][1024] bf16
    const unsigned short* __restrict__ wih,  // [2][1536][1024] bf16
    const float* __restrict__ bih,           // [2][1536]
    float* __restrict__ gx) {                // [2][2048][1536]
  int wid  = blockIdx.x * 8 + (threadIdx.x >> 5);  // 0..6143
  int lane = threadIdx.x & 31;
  int ln = lane & 15, hi = lane >> 4;
  int dir = wid / (128 * 24);
  int rem = wid % (128 * 24);
  int mt = rem / 24, ng = rem % 24;                // ng = group of 4 N-tiles
  const unsigned short* arow = x + (size_t)(16 * mt + ln) * 1024;
  const unsigned short* brow = wih + (size_t)dir * 1536 * 1024 +
                               (size_t)(64 * ng + ln) * 1024;
  f32x8 acc[4] = {};
  // prologue: fragments for k0 = 0
  bf16x16 a = load_frag(arow, hi, 0);
  bf16x16 b[4];
#pragma unroll
  for (int j = 0; j < 4; ++j) b[j] = load_frag(brow + (size_t)j * 16 * 1024, hi, 0);
  // steady state: issue next K-step's loads before this K-step's WMMAs
#pragma unroll 4
  for (int k0 = 0; k0 < 1024 - 32; k0 += 32) {
    bf16x16 an = load_frag(arow, hi, k0 + 32);
    bf16x16 bn[4];
#pragma unroll
    for (int j = 0; j < 4; ++j)
      bn[j] = load_frag(brow + (size_t)j * 16 * 1024, hi, k0 + 32);
#pragma unroll
    for (int j = 0; j < 4; ++j)
      acc[j] = __builtin_amdgcn_wmma_f32_16x16x32_bf16(false, a, false, b[j],
                                                       (short)0, acc[j],
                                                       false, false);
    a = an;
#pragma unroll
    for (int j = 0; j < 4; ++j) b[j] = bn[j];
  }
#pragma unroll
  for (int j = 0; j < 4; ++j)
    acc[j] = __builtin_amdgcn_wmma_f32_16x16x32_bf16(false, a, false, b[j],
                                                     (short)0, acc[j],
                                                     false, false);
  const float* bih_d = bih + dir * 1536;
  float* gxd = gx + (size_t)dir * 2048 * 1536;
#pragma unroll
  for (int j = 0; j < 4; ++j) {
    int n = 64 * ng + 16 * j + ln;
    float bb = bih_d[n];
#pragma unroll
    for (int r = 0; r < 8; ++r) {
      int m = 16 * mt + r + 8 * hi;
      gxd[(size_t)m * 1536 + n] = acc[j][r] + bb;
    }
  }
}

// ---------- stage 2b: GRU recurrence (persistent; 1 block per direction) -----
// LDS: gh[32][1536] f32 (192KB) + h f32 (64KB) + h bf16 (32KB) = 288KB/WGP.
// Wave w owns M-tile (w>>4) and 6 consecutive N-tiles. Per K-step, the LDS h
// fragment and all 6 whh fragments are loaded up front, then the 6 WMMAs run
// back-to-back (no per-WMMA full loadcnt drain on the serial critical path).

__global__ void __launch_bounds__(1024) gru_recur_kernel(
    const unsigned short* __restrict__ whh,  // [2][1536][512] bf16 (layer slice)
    const float* __restrict__ bhh,           // [2][1536]
    const float* __restrict__ gx,            // [2][2048][1536]
    float* __restrict__ out_f32,             // [2048][1024] (dir half) or null
    unsigned short* __restrict__ out_b16) {  // [2048][1024] (dir half) or null
  extern __shared__ char smem[];
  float* gh = (float*)smem;                                          // 32*1536
  float* hf = (float*)(smem + 32 * 1536 * 4);                        // 32*512
  unsigned short* hb = (unsigned short*)(smem + 32 * 1536 * 4 + 32 * 512 * 4);
  int dir = blockIdx.x;
  int tid = threadIdx.x;
  const unsigned short* whh_d = whh + (size_t)dir * 1536 * 512;
  const float* bhh_d = bhh + dir * 1536;
  const float* gx_d  = gx + (size_t)dir * 2048 * 1536;

  for (int e = tid; e < 32 * 512; e += 1024) { hf[e] = 0.f; hb[e] = 0; }
  __syncthreads();

  int wave = tid >> 5, lane = tid & 31, ln = lane & 15, hi = lane >> 4;
  int mt  = wave >> 4;            // 0..1
  int ntb = (wave & 15) * 6;      // base N-tile, 6 tiles per wave
  const unsigned short* arow = hb + (16 * mt + ln) * 512;

  for (int step = 0; step < 64; ++step) {
    int s = dir ? (63 - step) : step;
    // gh = h @ whh^T : K=512, A reused across 6 N-tiles
    f32x8 acc[6] = {};
#pragma unroll 2
    for (int k0 = 0; k0 < 512; k0 += 32) {
      bf16x16 a = load_frag(arow, hi, k0);
      bf16x16 b[6];
#pragma unroll
      for (int i = 0; i < 6; ++i)
        b[i] = load_frag(whh_d + (size_t)(16 * (ntb + i) + ln) * 512, hi, k0);
#pragma unroll
      for (int i = 0; i < 6; ++i)
        acc[i] = __builtin_amdgcn_wmma_f32_16x16x32_bf16(false, a, false, b[i],
                                                         (short)0, acc[i],
                                                         false, false);
    }
#pragma unroll
    for (int i = 0; i < 6; ++i) {
      int n = 16 * (ntb + i) + ln;
#pragma unroll
      for (int r = 0; r < 8; ++r) {
        int m = 16 * mt + r + 8 * hi;
        gh[m * 1536 + n] = acc[i][r];
      }
    }
    __syncthreads();
    // fused gate math; each thread owns 16 (b, j) elements
    int snext = dir ? (62 - step) : (step + 1);
#pragma unroll
    for (int q = 0; q < 16; ++q) {
      int e = tid + q * 1024;
      int b = e >> 9, j = e & 511;
      size_t tok = (size_t)(b * 64 + s);
      const float* gxr = gx_d + tok * 1536;
      float xr = gxr[j], xz = gxr[512 + j], xn = gxr[1024 + j];
      float hr = gh[b * 1536 + j]        + bhh_d[j];
      float hz = gh[b * 1536 + 512 + j]  + bhh_d[512 + j];
      float hn = gh[b * 1536 + 1024 + j] + bhh_d[1024 + j];
      float r  = 1.f / (1.f + __expf(-(xr + hr)));
      float z  = 1.f / (1.f + __expf(-(xz + hz)));
      float nn = tanhf(xn + r * hn);
      float hnew = (1.f - z) * nn + z * hf[e];
      hf[e] = hnew;
      hb[e] = f2bf(hnew);
      if (out_f32) out_f32[tok * 1024 + dir * 512 + j] = hnew;
      if (out_b16) out_b16[tok * 1024 + dir * 512 + j] = f2bf(hnew);
      // prefetch next step's gx rows (overlaps with next WMMA phase)
      if (step < 63)
        __builtin_prefetch(gx_d + (size_t)(b * 64 + snext) * 1536 + j, 0, 0);
    }
    __syncthreads();
  }
}

// ---------- stage 3: attention head + linear + softmax -----------------------

__global__ void __launch_bounds__(256) head_kernel(
    const float* __restrict__ states,    // [32][64][1024]
    const float* __restrict__ demoip,    // [32][3]
    const float* __restrict__ conv2_w,   // [1024]
    const float* __restrict__ conv2_b,   // [1]
    const float* __restrict__ lin_w,     // [2][1027]
    const float* __restrict__ lin_b,     // [2]
    float* __restrict__ out,             // [32][2]
    float* __restrict__ ctx_out,         // [32][1024]
    float* __restrict__ alpha_out) {     // [32][64]
  __shared__ float red[256];
  __shared__ float sdot[64];
  __shared__ float ctx[1024];
  __shared__ float smax, ssum;
  __shared__ float logit[2];
  int b = blockIdx.x, tid = threadIdx.x;
  const float* st = states + (size_t)b * 64 * 1024;
  for (int s = 0; s < 64; ++s) {
    float p = 0.f;
    for (int j = tid; j < 1024; j += 256) p += st[s * 1024 + j] * conv2_w[j];
    red[tid] = p; __syncthreads();
    for (int o = 128; o > 0; o >>= 1) { if (tid < o) red[tid] += red[tid + o]; __syncthreads(); }
    if (tid == 0) sdot[s] = red[0] + conv2_b[0];
    __syncthreads();
  }
  if (tid == 0) {
    float m = -3.4e38f;
    for (int s = 0; s < 64; ++s) m = fmaxf(m, sdot[s]);
    float sum = 0.f;
    for (int s = 0; s < 64; ++s) sum += __expf(sdot[s] - m);
    smax = m; ssum = sum;
  }
  __syncthreads();
  if (tid < 64) {
    float a = __expf(sdot[tid] - smax) / ssum;
    sdot[tid] = a;
    alpha_out[b * 64 + tid] = a;
  }
  __syncthreads();
  for (int j = tid; j < 1024; j += 256) {
    float c = 0.f;
    for (int s = 0; s < 64; ++s) c += sdot[s] * st[s * 1024 + j];
    ctx[j] = c;
    ctx_out[(size_t)b * 1024 + j] = c;
  }
  __syncthreads();
  for (int o = 0; o < 2; ++o) {
    float p = 0.f;
    for (int j = tid; j < 1027; j += 256) {
      float v = (j < 1024) ? ctx[j] : demoip[b * 3 + (j - 1024)];
      p += lin_w[o * 1027 + j] * v;
    }
    red[tid] = p; __syncthreads();
    for (int q = 128; q > 0; q >>= 1) { if (tid < q) red[tid] += red[tid + q]; __syncthreads(); }
    if (tid == 0) logit[o] = red[0] + lin_b[o];
    __syncthreads();
  }
  if (tid == 0) {
    float m = fmaxf(logit[0], logit[1]);
    float e0 = __expf(logit[0] - m), e1 = __expf(logit[1] - m);
    out[b * 2 + 0] = e0 / (e0 + e1);
    out[b * 2 + 1] = e1 / (e0 + e1);
  }
}

// ---------- launcher ---------------------------------------------------------

extern "C" void kernel_launch(void* const* d_in, const int* in_sizes, int n_in,
                              void* d_out, int out_size, void* d_ws, size_t ws_size,
                              hipStream_t stream) {
  (void)in_sizes; (void)n_in; (void)out_size; (void)ws_size;
  const float* inputs  = (const float*)d_in[0];
  const float* demoip  = (const float*)d_in[1];
  const float* conv_w  = (const float*)d_in[2];
  const float* conv_b  = (const float*)d_in[3];
  const float* conv2_w = (const float*)d_in[4];
  const float* conv2_b = (const float*)d_in[5];
  const float* wih     = (const float*)d_in[6];
  const float* whh     = (const float*)d_in[7];
  const float* bih     = (const float*)d_in[8];
  const float* bhh     = (const float*)d_in[9];
  const float* lin_w   = (const float*)d_in[10];
  const float* lin_b   = (const float*)d_in[11];

  // workspace layout (bytes)
  char* ws = (char*)d_ws;
  unsigned short* emb  = (unsigned short*)(ws + 0);          // 2048*1024 bf16 (4 MB)
  unsigned short* x1   = (unsigned short*)(ws + 4194304);    // 2048*1024 bf16 (4 MB)
  unsigned short* wihb = (unsigned short*)(ws + 8388608);    // 2*2*1536*1024 bf16 (12 MB)
  unsigned short* whhb = (unsigned short*)(ws + 20971520);   // 2*2*1536*512 bf16 (6 MB)
  float*          gx   = (float*)(ws + 27262976);            // 2*2048*1536 f32 (24 MB)

  // d_out layout (floats): out | conv_wts | states | context | alpha
  float* out_f   = (float*)d_out;
  float* cw_out  = out_f + 64;
  float* st_out  = out_f + 262208;
  float* ctx_out = out_f + 2359360;
  float* al_out  = out_f + 2392128;

  // cast GRU weights to bf16 once per call (deterministic)
  cast_bf16_kernel<<<24576, 256, 0, stream>>>(wih, wihb, 6291456);
  cast_bf16_kernel<<<12288, 256, 0, stream>>>(whh, whhb, 3145728);

  attn_conv_kernel<<<2048, 256, 0, stream>>>(inputs, conv_w, conv_b, cw_out, emb);

  for (int l = 0; l < 2; ++l) {
    const unsigned short* xin = (l == 0) ? emb : x1;
    gemm_gx_kernel<<<768, 256, 0, stream>>>(xin, wihb + (size_t)l * 2 * 1536 * 1024,
                                            bih + l * 2 * 1536, gx);
    gru_recur_kernel<<<2, 1024, 294912, stream>>>(
        whhb + (size_t)l * 2 * 1536 * 512, bhh + l * 2 * 1536, gx,
        (l == 1) ? st_out : nullptr,   // layer-1 states -> output
        (l == 0) ? x1 : nullptr);      // layer-0 states -> next-layer bf16 input
  }

  head_kernel<<<32, 256, 0, stream>>>(st_out, demoip, conv2_w, conv2_b,
                                      lin_w, lin_b, out_f, ctx_out, al_out);
}
